// CausalAttention_3169685864468
// MI455X (gfx1250) — compile-verified, matrix operand order checked by reading, not run on hardware
//
#include <hip/hip_runtime.h>
#include <hip/hip_bf16.h>

typedef __attribute__((ext_vector_type(16))) __bf16 v16bf;
typedef __attribute__((ext_vector_type(8)))  __bf16 v8bf;
typedef __attribute__((ext_vector_type(4)))  __bf16 v4bf;
typedef __attribute__((ext_vector_type(8)))  float  v8f;
typedef __attribute__((ext_vector_type(4)))  int    v4i;
typedef __attribute__((ext_vector_type(8)))  int    v8i;
typedef __attribute__((ext_vector_type(4)))  unsigned int v4u;

#define BATCH   2
#define SEQ     2048
#define DMODEL  2048
#define NHEADS  16
#define HDIM    128
#define MTOT    (BATCH * SEQ)

// ---------------------------------------------------------------------------
// CDNA5 async global->LDS (ASYNCcnt) with graceful fallback.
// Toolchain prototype (from diagnostics): (v4i AS1*, v4i AS3*, imm, imm).
// ---------------------------------------------------------------------------
#if __has_builtin(__builtin_amdgcn_global_load_async_to_lds_b128)
#define ASYNC_LDS 1
#else
#define ASYNC_LDS 0
#endif

// Tensor Data Mover (TENSORcnt) -- this toolchain: 6-arg prototype
// (uint32x4 g0, int32x8 g1, int32x4 g2, int32x4 g3, int32x8, i32 cpol)
#if __has_builtin(__builtin_amdgcn_tensor_load_to_lds) && \
    __has_builtin(__builtin_amdgcn_s_wait_tensorcnt)
#define HAVE_TDM 1
#else
#define HAVE_TDM 0
#endif

__device__ __forceinline__ void cp16_g2l(__bf16* dst_lds, const __bf16* src_g) {
#if ASYNC_LDS
  v4i* s0 = (v4i*)src_g;    // reinterpret (drops const)
  v4i* d0 = (v4i*)dst_lds;  // reinterpret
  __builtin_amdgcn_global_load_async_to_lds_b128(
      (__attribute__((address_space(1))) v4i*)s0,
      (__attribute__((address_space(3))) v4i*)d0,
      /*offset=*/0, /*cpol=*/0);
#else
  *(v8bf*)dst_lds = *(const v8bf*)src_g;
#endif
}

__device__ __forceinline__ void wait_async_copies() {
#if ASYNC_LDS && __has_builtin(__builtin_amdgcn_s_wait_asynccnt)
  __builtin_amdgcn_s_wait_asynccnt(0);
#endif
  // fallback: compiler-inserted loadcnt/dscnt waits before the barrier
}

// LDS byte offset of a __shared__ object (ptrtoint of the AS(3) pointer)
__device__ __forceinline__ unsigned lds_offset_of(const __bf16* p) {
  return (unsigned)(size_t)(__attribute__((address_space(3))) const __bf16*)p;
}

// ---------------------------------------------------------------------------
// TDM: DMA a 2D tile (tile_y rows x tile_x bf16) from a row-major tensor
// (row stride = stride_elems) into LDS with 16B row padding (LDT = tile_x+8).
// D# layout per CDNA5 ISA ch.8: group0 = {flags, lds_addr, global_addr, type},
// group1 = {mask/size/pad, dims, tile dims, strides}; groups 2/3 zero (2-D).
// ---------------------------------------------------------------------------
__device__ __forceinline__ void tdm_load_2d(unsigned lds_off,
                                            const __bf16* gptr,
                                            unsigned tile_x, unsigned tile_y,
                                            unsigned stride_elems) {
#if HAVE_TDM
  unsigned long long ga = (unsigned long long)(size_t)gptr;
  v4u g0;
  g0[0] = 1u;                                        // count=1 (valid, user)
  g0[1] = lds_off;                                   // lds_addr (bytes)
  g0[2] = (unsigned)(ga & 0xffffffffu);              // global_addr[31:0]
  g0[3] = (unsigned)((ga >> 32) & 0x01ffffffu)       // global_addr[56:32]
          | (2u << 30);                              // type = 2 ("image")
  // data_size=1 (2B), pad_enable, pad_interval code 4 (=32 DW = 128B rows),
  // pad_amount code 3 (=4 DW = 16B): matches LDT = tile_x + 8 bf16.
  v8i g1;
  g1[0] = (int)((1u << 16) | (1u << 20) | (4u << 22) | (3u << 25));
  g1[1] = 0;                                         // dim0[15:0]=0 (2^30)
  g1[2] = (int)0x4000u;                              // dim0[31:16]; dim1 lo=0
  g1[3] = (int)(0x4000u | (tile_x << 16));           // dim1 hi; tile_dim0
  g1[4] = (int)tile_y;                               // tile_dim1; tile_dim2=0
  g1[5] = (int)stride_elems;                         // dim0_stride[31:0]
  g1[6] = 0;                                         // stride hi; dim1_stride
  g1[7] = 0;
  v4i z4 = {0, 0, 0, 0};
  v8i z8 = {0, 0, 0, 0, 0, 0, 0, 0};
  __builtin_amdgcn_tensor_load_to_lds(g0, g1, z4, z4, z8, /*cpol=*/0);
#else
  (void)lds_off; (void)gptr; (void)tile_x; (void)tile_y; (void)stride_elems;
#endif
}

// ---------------------------------------------------------------------------
// WMMA helper: D = A(16x32 bf16) x B(32x16 bf16) + C(16x16 f32)
// ---------------------------------------------------------------------------
__device__ __forceinline__ v8f wmma_bf16(v16bf a, v16bf b, v8f c) {
  return __builtin_amdgcn_wmma_f32_16x16x32_bf16(
      /*neg_a=*/false, a, /*neg_b=*/false, b,
      /*c_mod=*/(short)0, c, /*reuse_a=*/false, /*reuse_b=*/false);
}

// Load a 16x32 A-operand (or symmetric B-operand) fragment from a row-major
// bf16 matrix. Lanes 0-15 hold rows r0..r0+15 with K in [k0,k0+8) then
// [k0+16,k0+24); lanes 16-31 hold K offsets +8.
__device__ __forceinline__ v16bf load_frag(const __bf16* base, int ld,
                                           int r0, int k0, int lane) {
  const int half = lane >> 4, lr = lane & 15;
  const __bf16* p = base + (r0 + lr) * ld + k0 + half * 8;
  union { v16bf f; v8bf h[2]; } u;
  u.h[0] = *(const v8bf*)(p);
  u.h[1] = *(const v8bf*)(p + 16);
  return u.f;
}

// ---------------------------------------------------------------------------
// fp32 -> bf16 conversion (vectorized)
// ---------------------------------------------------------------------------
__global__ void cvt_f32_bf16(const float* __restrict__ in,
                             __bf16* __restrict__ out, int n4) {
  int i = blockIdx.x * blockDim.x + threadIdx.x;
  if (i < n4) {
    float4 v = ((const float4*)in)[i];
    v4bf o;
    o[0] = (__bf16)v.x; o[1] = (__bf16)v.y;
    o[2] = (__bf16)v.z; o[3] = (__bf16)v.w;
    ((v4bf*)out)[i] = o;
  }
}

// ---------------------------------------------------------------------------
// Y[M,N] = A[M,K](bf16) @ W[N,K]^T(bf16) + bias[N](f32)
// Block: 256 threads = 8 waves, 128x64 tile, wave = 32x32; LDS double-buffered.
// Tiles are DMA'd by the Tensor Data Mover (wave 0 issues, TENSORcnt waits),
// falling back to per-thread async global->LDS copies.
// ---------------------------------------------------------------------------
template <bool OUT_BF16>
__global__ __launch_bounds__(256) void gemm_bias(
    const __bf16* __restrict__ A, const __bf16* __restrict__ W,
    const float* __restrict__ bias, void* __restrict__ Cout,
    int M, int N, int K) {
  constexpr int BM = 128, BN = 64, BK = 64, LDT = BK + 8;
  __shared__ __bf16 As[2][BM * LDT];   // 2 x 18432 B
  __shared__ __bf16 Ws[2][BN * LDT];   // 2 x  9216 B  (total 55296 B)

  const int tid = threadIdx.x;
  const int lane = tid & 31, wid = tid >> 5;
  const int wm = wid & 3, wn = wid >> 2;       // 4 waves along M, 2 along N
  const int half = lane >> 4, lr = lane & 15;
  const int m0 = blockIdx.y * BM, n0 = blockIdx.x * BN;

  const int arow = tid >> 1, acc_ = (tid & 1) * 32;   // A: 2 thr/row, 32 elems
  const int wrow = tid >> 2, wcc = (tid & 3) * 16;    // W: 4 thr/row, 16 elems

#if HAVE_TDM
  const unsigned ldsA = lds_offset_of(&As[0][0]);
  const unsigned ldsW = lds_offset_of(&Ws[0][0]);
#endif

  auto load_tiles = [&](int k0, int buf) {
#if HAVE_TDM
    if (wid == 0) {   // one wave issues the DMA descriptors (EXEC ignored)
      tdm_load_2d(ldsA + (unsigned)buf * BM * LDT * 2,
                  A + (size_t)m0 * K + k0, BK, BM, K);
      tdm_load_2d(ldsW + (unsigned)buf * BN * LDT * 2,
                  W + (size_t)n0 * K + k0, BK, BN, K);
    }
#else
    const __bf16* sa = A + (size_t)(m0 + arow) * K + k0 + acc_;
    __bf16* da = &As[buf][arow * LDT + acc_];
    cp16_g2l(da, sa); cp16_g2l(da + 8, sa + 8);
    cp16_g2l(da + 16, sa + 16); cp16_g2l(da + 24, sa + 24);
    const __bf16* sw = W + (size_t)(n0 + wrow) * K + k0 + wcc;
    __bf16* dw = &Ws[buf][wrow * LDT + wcc];
    cp16_g2l(dw, sw); cp16_g2l(dw + 8, sw + 8);
#endif
  };
  auto wait_tiles = [&]() {
#if HAVE_TDM
    if (wid == 0) __builtin_amdgcn_s_wait_tensorcnt(0);
#else
    wait_async_copies();
#endif
  };

  v8f acc[2][2] = {};

  load_tiles(0, 0);
  wait_tiles();
  __syncthreads();

  int cur = 0;
  for (int k0 = 0; k0 < K; k0 += BK) {
    if (k0 + BK < K) load_tiles(k0 + BK, cur ^ 1);   // overlaps with WMMAs
#pragma unroll
    for (int kk = 0; kk < BK; kk += 32) {
      v16bf a0 = load_frag(As[cur], LDT, wm * 32,      kk, lane);
      v16bf a1 = load_frag(As[cur], LDT, wm * 32 + 16, kk, lane);
      v16bf b0 = load_frag(Ws[cur], LDT, wn * 32,      kk, lane);
      v16bf b1 = load_frag(Ws[cur], LDT, wn * 32 + 16, kk, lane);
      acc[0][0] = wmma_bf16(a0, b0, acc[0][0]);
      acc[0][1] = wmma_bf16(a0, b1, acc[0][1]);
      acc[1][0] = wmma_bf16(a1, b0, acc[1][0]);
      acc[1][1] = wmma_bf16(a1, b1, acc[1][1]);
    }
    wait_tiles();
    __syncthreads();
    cur ^= 1;
  }

#pragma unroll
  for (int nt = 0; nt < 2; ++nt) {
    int n = n0 + wn * 32 + nt * 16 + lr;
    float bv = bias[n];
#pragma unroll
    for (int mt = 0; mt < 2; ++mt) {
#pragma unroll
      for (int r = 0; r < 8; ++r) {
        int m = m0 + wm * 32 + mt * 16 + r + 8 * half;
        float v = acc[mt][nt][r] + bv;
        if (OUT_BF16)
          ((__bf16*)Cout)[(size_t)m * N + n] = (__bf16)v;
        else
          ((float*)Cout)[(size_t)m * N + n] = v;
      }
    }
  }
}

// ---------------------------------------------------------------------------
// Flash attention (causal), bf16 compute / f32 online softmax.
// grid = (SEQ/128, BATCH*NHEADS); block = 256 (8 waves, 16 query rows each).
// ---------------------------------------------------------------------------
__global__ __launch_bounds__(256) void flash_attn(
    const __bf16* __restrict__ Q, const __bf16* __restrict__ Kc,
    const __bf16* __restrict__ Vc, __bf16* __restrict__ Oc) {
  constexpr int BR = 128, BKEY = 64;
  constexpr int LDK = HDIM + 8;   // 136
  constexpr int LDV = BKEY + 8;   // 72 (V stored transposed: [d][key])
  constexpr int LDP = BKEY + 8;   // 72
  __shared__ __bf16 Ks[BKEY * LDK];
  __shared__ __bf16 Vts[HDIM * LDV];
  __shared__ __bf16 Ps[8 * 16 * LDP];

  const int tid = threadIdx.x, lane = tid & 31, wid = tid >> 5;
  const int half = lane >> 4, lr = lane & 15;
  const int bh = blockIdx.y;
  const int b = bh / NHEADS, h = bh % NHEADS;
  const int q0 = blockIdx.x * BR;
  const int qw = q0 + wid * 16;          // this wave's 16 query rows
  const float scale = 0.08838834764831845f;  // 1/sqrt(128)

  const size_t hoff = (size_t)h * HDIM;
  const __bf16* Qb = Q  + (size_t)b * SEQ * DMODEL + hoff;
  const __bf16* Kb = Kc + (size_t)b * SEQ * DMODEL + hoff;
  const __bf16* Vb = Vc + (size_t)b * SEQ * DMODEL + hoff;

  // Q fragments for this wave: 4 chunks of K=32 over hd=128
  v16bf qf[4];
#pragma unroll
  for (int c = 0; c < 4; ++c)
    qf[c] = load_frag(Qb + (size_t)qw * DMODEL, DMODEL, 0, c * 32, lane);

  v8f o[8] = {};
  float mrun[8], lrun[8];
#pragma unroll
  for (int r = 0; r < 8; ++r) { mrun[r] = -1e30f; lrun[r] = 0.f; }

  const int jmax = (q0 + BR) / BKEY;     // causal: only keys <= q0+127
  const int krow = tid >> 2, kcc = (tid & 3) * 32;  // cooperative tile mapping

  for (int j = 0; j < jmax; ++j) {
    const int kb = j * BKEY;

    // K tile 64x128 row-major: async, overlaps with the V transpose below
    {
      const __bf16* src = Kb + (size_t)(kb + krow) * DMODEL + kcc;
      __bf16* dst = Ks + krow * LDK + kcc;
      cp16_g2l(dst, src); cp16_g2l(dst + 8, src + 8);
      cp16_g2l(dst + 16, src + 16); cp16_g2l(dst + 24, src + 24);
    }
    // V tile stored transposed [d][key]: gather loads first, then scatter
    {
      const __bf16* src = Vb + (size_t)(kb + krow) * DMODEL + kcc;
      v8bf vv[4];
#pragma unroll
      for (int u = 0; u < 4; ++u) vv[u] = *(const v8bf*)(src + u * 8);
#pragma unroll
      for (int u = 0; u < 4; ++u)
#pragma unroll
        for (int e = 0; e < 8; ++e)
          Vts[(kcc + u * 8 + e) * LDV + krow] = vv[u][e];
    }
    // prefetch next key block while this one is consumed
    if (j + 1 < jmax) {
      __builtin_prefetch(Kb + (size_t)(kb + BKEY + krow) * DMODEL + kcc, 0, 1);
      __builtin_prefetch(Vb + (size_t)(kb + BKEY + krow) * DMODEL + kcc, 0, 1);
    }
    wait_async_copies();
    __syncthreads();

    // scores: S[16q x 64k] = Q @ K^T   (4 n-tiles x 4 K-chunks)
    v8f s[4] = {};
#pragma unroll
    for (int nt = 0; nt < 4; ++nt)
#pragma unroll
      for (int c = 0; c < 4; ++c) {
        v16bf kf = load_frag(Ks, LDK, nt * 16, c * 32, lane);
        s[nt] = wmma_bf16(qf[c], kf, s[nt]);
      }

    // scale + causal mask (element row = r + 8*half, col = kb + nt*16 + lr)
#pragma unroll
    for (int nt = 0; nt < 4; ++nt) {
      int kcol = kb + nt * 16 + lr;
#pragma unroll
      for (int r = 0; r < 8; ++r) {
        float v = s[nt][r] * scale;
        int qrow = qw + r + 8 * half;
        s[nt][r] = (kcol > qrow) ? -1e30f : v;
      }
    }

    // online softmax; 16-lane xor reductions align with C/D layout
    float pl[4][8];
#pragma unroll
    for (int r = 0; r < 8; ++r) {
      float mx = fmaxf(fmaxf(s[0][r], s[1][r]), fmaxf(s[2][r], s[3][r]));
#pragma unroll
      for (int d = 1; d < 16; d <<= 1) mx = fmaxf(mx, __shfl_xor(mx, d, 32));
      float mnew = fmaxf(mrun[r], mx);
      float alpha = __expf(mrun[r] - mnew);
      float rs = 0.f;
#pragma unroll
      for (int nt = 0; nt < 4; ++nt) {
        float p = __expf(s[nt][r] - mnew);
        pl[nt][r] = p; rs += p;
      }
#pragma unroll
      for (int d = 1; d < 16; d <<= 1) rs += __shfl_xor(rs, d, 32);
      lrun[r] = lrun[r] * alpha + rs;
      mrun[r] = mnew;
#pragma unroll
      for (int dt = 0; dt < 8; ++dt) o[dt][r] *= alpha;
    }

    // P (D-layout) -> LDS -> A-operand layout (per-wave region, in-order DS)
    __bf16* pp = Ps + wid * 16 * LDP;
#pragma unroll
    for (int nt = 0; nt < 4; ++nt)
#pragma unroll
      for (int r = 0; r < 8; ++r)
        pp[(r + 8 * half) * LDP + nt * 16 + lr] = (__bf16)pl[nt][r];

    // O += P @ V (2 K-chunks over 64 keys, 8 d-tiles)
#pragma unroll
    for (int c2 = 0; c2 < 2; ++c2) {
      v16bf pa = load_frag(pp, LDP, 0, c2 * 32, lane);
#pragma unroll
      for (int dt = 0; dt < 8; ++dt) {
        v16bf vb = load_frag(Vts, LDV, dt * 16, c2 * 32, lane);
        o[dt] = wmma_bf16(pa, vb, o[dt]);
      }
    }
    __syncthreads();
  }

  // normalize and store bf16 attn output [B,S,D]
#pragma unroll
  for (int r = 0; r < 8; ++r) {
    float inv = 1.0f / lrun[r];
    int qrow = qw + r + 8 * half;
    __bf16* orow = Oc + ((size_t)b * SEQ + qrow) * DMODEL + hoff;
#pragma unroll
    for (int dt = 0; dt < 8; ++dt)
      orow[dt * 16 + lr] = (__bf16)(o[dt][r] * inv);
  }
}

// ---------------------------------------------------------------------------
// Host launcher
// ---------------------------------------------------------------------------
extern "C" void kernel_launch(void* const* d_in, const int* in_sizes, int n_in,
                              void* d_out, int out_size, void* d_ws,
                              size_t ws_size, hipStream_t stream) {
  const float* x  = (const float*)d_in[0];
  const float* wq = (const float*)d_in[1];
  const float* bq = (const float*)d_in[2];
  const float* wk = (const float*)d_in[3];
  const float* bk = (const float*)d_in[4];
  const float* wv = (const float*)d_in[5];
  const float* bv = (const float*)d_in[6];
  const float* wo = (const float*)d_in[7];
  const float* bo = (const float*)d_in[8];
  float* out = (float*)d_out;

  char* ws = (char*)d_ws;
  size_t off = 0;
  auto alloc = [&](size_t bytes) -> void* {
    void* p = ws + off;
    off += (bytes + 255) & ~(size_t)255;
    return p;
  };
  const size_t nx = (size_t)MTOT * DMODEL;       // 8M elements
  const size_t nw = (size_t)DMODEL * DMODEL;     // 4M elements
  __bf16* xb  = (__bf16*)alloc(nx * 2);
  __bf16* wqb = (__bf16*)alloc(nw * 2);
  __bf16* wkb = (__bf16*)alloc(nw * 2);
  __bf16* wvb = (__bf16*)alloc(nw * 2);
  __bf16* wob = (__bf16*)alloc(nw * 2);
  __bf16* Qh  = (__bf16*)alloc(nx * 2);
  __bf16* Kh  = (__bf16*)alloc(nx * 2);
  __bf16* Vh  = (__bf16*)alloc(nx * 2);
  __bf16* Ah  = (__bf16*)alloc(nx * 2);

  auto cvt = [&](const float* src, __bf16* dst, size_t n) {
    int n4 = (int)(n / 4);
    cvt_f32_bf16<<<(n4 + 255) / 256, 256, 0, stream>>>(src, dst, n4);
  };
  cvt(x,  xb,  nx);
  cvt(wq, wqb, nw);
  cvt(wk, wkb, nw);
  cvt(wv, wvb, nw);
  cvt(wo, wob, nw);

  dim3 gg(DMODEL / 64, MTOT / 128);  // (32, 32)
  gemm_bias<true><<<gg, 256, 0, stream>>>(xb, wqb, bq, Qh, MTOT, DMODEL, DMODEL);
  gemm_bias<true><<<gg, 256, 0, stream>>>(xb, wkb, bk, Kh, MTOT, DMODEL, DMODEL);
  gemm_bias<true><<<gg, 256, 0, stream>>>(xb, wvb, bv, Vh, MTOT, DMODEL, DMODEL);

  dim3 ga(SEQ / 128, BATCH * NHEADS);  // (16, 32)
  flash_attn<<<ga, 256, 0, stream>>>(Qh, Kh, Vh, Ah);

  gemm_bias<false><<<gg, 256, 0, stream>>>(Ah, wob, bo, out, MTOT, DMODEL, DMODEL);
}
